// SelfAttentionLayer_52381421142698
// MI455X (gfx1250) — matrix operand order repaired and checked by reading, not compiled, not run on hardware
//
#include <hip/hip_runtime.h>

typedef __attribute__((ext_vector_type(16))) _Float16 v16h;
typedef __attribute__((ext_vector_type(8)))  _Float16 v8h;
typedef __attribute__((ext_vector_type(4)))  _Float16 v4h;
typedef __attribute__((ext_vector_type(8)))  float    v8f;
typedef __attribute__((ext_vector_type(4)))  float    v4f;
typedef __attribute__((ext_vector_type(4)))  unsigned int v4u;
typedef __attribute__((ext_vector_type(8)))  int      v8i;
typedef __attribute__((ext_vector_type(4)))  int      v4i;

#define N_BATCH 4
#define T_SEQ   4096
#define D_MODEL 1024
#define KV_DIM  1024

// LDS A-panel: 256 rows x 32 k-halves, padded by 8 halves (16B) per row by the
// TDM pad feature -> 80B row stride (20 DWORDs; gcd(20,64)=4 -> 16 consecutive
// rows land on distinct bank groups for the 16-lane ds_load_b128 pattern).
#define PANEL_ROWS   256
#define PANEL_K      32
#define LDS_STRIDE   40   /* halves: 32 data + 8 pad */

// ---------------- convert x: f32 -> f16, vectorized x4 ----------------
__global__ __launch_bounds__(256) void attn_cvt_x(const float* __restrict__ src,
                                                  _Float16* __restrict__ dst, int n4) {
  int i = blockIdx.x * blockDim.x + threadIdx.x;
  if (i < n4) {
    v4f v = *(const v4f*)(src + (size_t)i * 4);
    v4h h;
    h.x = (_Float16)v.x; h.y = (_Float16)v.y;
    h.z = (_Float16)v.z; h.w = (_Float16)v.w;
    *(v4h*)(dst + (size_t)i * 4) = h;
  }
}

// ------------- transpose+convert W (D x K f32) -> Wt (K x D f16) -------------
__global__ __launch_bounds__(256) void attn_transpose_w(
    const float* __restrict__ W0, const float* __restrict__ W1,
    const float* __restrict__ W2,
    _Float16* __restrict__ T0, _Float16* __restrict__ T1,
    _Float16* __restrict__ T2) {
  const float* W = (blockIdx.z == 0) ? W0 : (blockIdx.z == 1) ? W1 : W2;
  _Float16*   Tw = (blockIdx.z == 0) ? T0 : (blockIdx.z == 1) ? T1 : T2;
  __shared__ float tile[32][33];
  int bx = blockIdx.x * 32;
  int by = blockIdx.y * 32;
  int tx = threadIdx.x & 31;
  int ty = threadIdx.x >> 5;
  for (int r = ty; r < 32; r += 8)
    tile[r][tx] = W[(size_t)(by + r) * KV_DIM + bx + tx];
  __syncthreads();
  for (int r = ty; r < 32; r += 8)
    Tw[(size_t)(bx + r) * D_MODEL + by + tx] = (_Float16)tile[tx][r];
}

// ---- TDM: issue a 2D strided-tile load (PANEL_ROWS x PANEL_K f16) into LDS ----
// D# per CDNA5 ISA ch.8: group0 = {count/lds_addr/global_addr/type=2},
// group1 = {data_size=2B, pad(16B per 64B row), tensor dims, tile dims, stride}.
// This toolchain exposes the 6-arg builtin: (g0, g1, g2, g3, g_extra, cpol).
__device__ __forceinline__ void tdm_load_panel(const _Float16* gsrc, unsigned ldsAddr,
                                               int lda, int Kdim, int Mtot) {
  unsigned long long ga = (unsigned long long)(size_t)gsrc;
  v4u g0;
  g0[0] = 1u;                                               // count=1, user mode
  g0[1] = ldsAddr;                                          // LDS byte address
  g0[2] = (unsigned)(ga & 0xFFFFFFFFull);                   // global_addr[31:0]
  g0[3] = (unsigned)((ga >> 32) & 0x1FFFFFFull) | (2u << 30); // addr[56:32]|type=2
  v8i g1;
  long long st0 = (long long)lda;                           // dim0 stride (elems)
  g1[0] = (1 << 16)        /* data_size = 2 bytes */
        | (1 << 20)        /* pad_enable */
        | (3 << 22)        /* pad_interval: 16 DWORDs (one 64B row) */
        | (3 << 25);       /* pad_amount: 4 DWORDs (16B) */
  g1[1] = (Kdim & 0xFFFF) << 16;                            // tensor_dim0[15:0]
  g1[2] = ((unsigned)Kdim >> 16) | ((Mtot & 0xFFFF) << 16); // dim0 hi | dim1 lo
  g1[3] = ((unsigned)Mtot >> 16) | (PANEL_K << 16);         // dim1 hi | tile_dim0
  g1[4] = PANEL_ROWS;                                       // tile_dim1
  g1[5] = (int)(st0 & 0xFFFFFFFFll);                        // dim0_stride lo
  g1[6] = (int)((st0 >> 32) & 0xFFFFll);                    // dim0_stride hi
  g1[7] = 0;
  v4i z4 = {0, 0, 0, 0};                                    // 2D: groups 2/3 unused
  v8i z8 = {0, 0, 0, 0, 0, 0, 0, 0};
  __builtin_amdgcn_tensor_load_to_lds(g0, g1, z4, z4, z8, 0);
}

// ---------------- WMMA GEMM with TDM-staged A panel ----------------
// C[m,n] = scale * sum_k A[m,k] * Bt[n,k] (+ bias[n])
// A row-major MxK f16 (TDM -> LDS, double buffered); Bt row-major NxK f16.
// Workgroup: 8 waves, 256(M) x 128(N); wave: 32(M) x 128(N), 16 accumulators.
__global__ __launch_bounds__(256) void attn_gemm_wmma(
    const _Float16* __restrict__ A, int lda, long strideA,
    const _Float16* __restrict__ Bt, int ldb, long strideB,
    const float* __restrict__ bias, float scale,
    _Float16* __restrict__ outH, float* __restrict__ outF,
    int ldc, long strideC, int transC, int Kdim, int Mtot) {
  __shared__ _Float16 apanel[2][PANEL_ROWS * LDS_STRIDE];   // 40KB total

  int b = blockIdx.z;
  A  += (size_t)b * strideA;
  Bt += (size_t)b * strideB;
  if (outH) outH += (size_t)b * strideC;
  if (outF) outF += (size_t)b * strideC;

  int lane = threadIdx.x & 31;
  int wave = threadIdx.x >> 5;                 // 0..7 (wave32)
  int mBase = blockIdx.y * 256;                // workgroup M tile
  int m0 = mBase + wave * 32;                  // wave's 32 rows
  int n0 = blockIdx.x * 128;

  const _Float16* aPanelSrc = A + (size_t)mBase * lda;
  unsigned ldsA0 = (unsigned)(size_t)&apanel[0][0];
  unsigned ldsA1 = (unsigned)(size_t)&apanel[1][0];

  // A-frag (16x32): lane L -> local row (L&15), 16 contiguous halves
  //                 at K=(L>>4)*16.  B-frag (32x16): lane L -> col (L&15),
  //                 16 contiguous halves at K=(L>>4)*16.
  int rowL  = wave * 32 + (lane & 15);
  int koffA = (lane >> 4) * 16;
  int colB  = lane & 15;
  int koffB = (lane >> 4) * 16;

  v8f acc[2][8] = {};

  int nkb = Kdim / PANEL_K;
  if (wave == 0)
    tdm_load_panel(aPanelSrc, ldsA0, lda, Kdim, Mtot);

  for (int kb = 0; kb < nkb; ++kb) {
    int buf = kb & 1;
    if (wave == 0)
      __builtin_amdgcn_s_wait_tensorcnt(0);    // panel for this kb landed
    __syncthreads();                           // (also fences reads of buf^1)
    if (wave == 0 && kb + 1 < nkb)
      tdm_load_panel(aPanelSrc + (size_t)(kb + 1) * PANEL_K,
                     buf ? ldsA0 : ldsA1, lda, Kdim, Mtot);

    const _Float16* ap = &apanel[buf][0];
    v16h a0, a1;
    {
      const _Float16* p0 = ap + (size_t)rowL * LDS_STRIDE + koffA;
      *(v8h*)&a0       = *(const v8h*)p0;          // ds_load_b128
      *((v8h*)&a0 + 1) = *(const v8h*)(p0 + 8);
      const _Float16* p1 = p0 + 16 * LDS_STRIDE;   // rows +16
      *(v8h*)&a1       = *(const v8h*)p1;
      *((v8h*)&a1 + 1) = *(const v8h*)(p1 + 8);
    }
    int k0 = kb * PANEL_K;
#pragma unroll
    for (int j = 0; j < 8; ++j) {
      const _Float16* bp = Bt + (size_t)(n0 + j * 16 + colB) * ldb + k0 + koffB;
      v16h bfrag = *(const v16h*)bp;               // 2x global_load_b128
      acc[0][j] = __builtin_amdgcn_wmma_f32_16x16x32_f16(
          false, a0, false, bfrag, (short)0, acc[0][j], false, false);
      acc[1][j] = __builtin_amdgcn_wmma_f32_16x16x32_f16(
          false, a1, false, bfrag, (short)0, acc[1][j], false, false);
    }
  }

  // C/D layout: lane L, vgpr g -> M = base + g + 8*(L>>4), N = j*16 + (L&15)
  int mrowBase = m0 + 8 * (lane >> 4);
  int ncol0 = n0 + colB;
#pragma unroll
  for (int s = 0; s < 2; ++s) {
#pragma unroll
    for (int j = 0; j < 8; ++j) {
      int n = ncol0 + j * 16;
      float bv = bias ? bias[n] : 0.0f;
#pragma unroll
      for (int g = 0; g < 8; ++g) {
        int m = mrowBase + s * 16 + g;
        float val = acc[s][j][g] * scale + bv;
        size_t idx = transC ? ((size_t)n * ldc + m) : ((size_t)m * ldc + n);
        if (outH) outH[idx] = (_Float16)val;
        else      outF[idx] = val;
      }
    }
  }
}

// ---------------- row softmax over T_SEQ, in place on f16 ----------------
__global__ __launch_bounds__(256) void attn_softmax_rows(_Float16* __restrict__ S) {
  __shared__ float red[256];
  _Float16* p = S + (size_t)blockIdx.x * T_SEQ;
  int tid = threadIdx.x;
  float vals[16];
  v8h a0 = ((const v8h*)(p + tid * 16))[0];
  v8h a1 = ((const v8h*)(p + tid * 16))[1];
#pragma unroll
  for (int i = 0; i < 8; ++i) { vals[i] = (float)a0[i]; vals[i + 8] = (float)a1[i]; }

  float mx = vals[0];
#pragma unroll
  for (int i = 1; i < 16; ++i) mx = fmaxf(mx, vals[i]);
  red[tid] = mx;
  __syncthreads();
  for (int s = 128; s > 0; s >>= 1) {
    if (tid < s) red[tid] = fmaxf(red[tid], red[tid + s]);
    __syncthreads();
  }
  mx = red[0];
  __syncthreads();

  float sum = 0.0f;
#pragma unroll
  for (int i = 0; i < 16; ++i) { vals[i] = expf(vals[i] - mx); sum += vals[i]; }
  red[tid] = sum;
  __syncthreads();
  for (int s = 128; s > 0; s >>= 1) {
    if (tid < s) red[tid] += red[tid + s];
    __syncthreads();
  }
  float inv = 1.0f / red[0];

  v8h o0, o1;
#pragma unroll
  for (int i = 0; i < 8; ++i) {
    o0[i] = (_Float16)(vals[i] * inv);
    o1[i] = (_Float16)(vals[i + 8] * inv);
  }
  ((v8h*)(p + tid * 16))[0] = o0;
  ((v8h*)(p + tid * 16))[1] = o1;
}

extern "C" void kernel_launch(void* const* d_in, const int* in_sizes, int n_in,
                              void* d_out, int out_size, void* d_ws, size_t ws_size,
                              hipStream_t stream) {
  (void)in_sizes; (void)n_in; (void)out_size; (void)ws_size;
  const float* x  = (const float*)d_in[0];
  const float* Wq = (const float*)d_in[1];
  const float* bq = (const float*)d_in[2];
  const float* Wk = (const float*)d_in[3];
  const float* bk = (const float*)d_in[4];
  const float* Wv = (const float*)d_in[5];
  const float* bv = (const float*)d_in[6];
  float* out = (float*)d_out;

  const long NT = (long)N_BATCH * T_SEQ;
  const long XE = NT * D_MODEL;
  const long WE = (long)D_MODEL * KV_DIM;
  const long SE = (long)N_BATCH * T_SEQ * T_SEQ;

  _Float16* ws  = (_Float16*)d_ws;
  size_t off = 0;
  _Float16* xh  = ws + off; off += XE;
  _Float16* wqt = ws + off; off += WE;
  _Float16* wkt = ws + off; off += WE;
  _Float16* wvt = ws + off; off += WE;
  _Float16* qh  = ws + off; off += XE;   // per batch: T x K row-major
  _Float16* kh  = ws + off; off += XE;   // per batch: T x K row-major
  _Float16* vth = ws + off; off += XE;   // per batch: V_DIM x T (transposed)
  _Float16* S   = ws + off; off += SE;   // per batch: T x T scores / probs

  int n4 = (int)(XE / 4);
  attn_cvt_x<<<n4 / 256, 256, 0, stream>>>(x, xh, n4);

  attn_transpose_w<<<dim3(KV_DIM / 32, D_MODEL / 32, 3), 256, 0, stream>>>(
      Wq, Wk, Wv, wqt, wkt, wvt);

  long sXD = (long)T_SEQ * D_MODEL;
  long sTT = (long)T_SEQ * T_SEQ;

  // QKV projections: per batch M=T(4096), N=KV_DIM(1024), K=D(1024)
  dim3 gProj(KV_DIM / 128, T_SEQ / 256, N_BATCH);
  attn_gemm_wmma<<<gProj, 256, 0, stream>>>(xh, D_MODEL, sXD, wqt, D_MODEL, 0,
      bq, 1.0f, qh, nullptr, KV_DIM, sXD, 0, D_MODEL, T_SEQ);
  attn_gemm_wmma<<<gProj, 256, 0, stream>>>(xh, D_MODEL, sXD, wkt, D_MODEL, 0,
      bk, 1.0f, kh, nullptr, KV_DIM, sXD, 0, D_MODEL, T_SEQ);
  attn_gemm_wmma<<<gProj, 256, 0, stream>>>(xh, D_MODEL, sXD, wvt, D_MODEL, 0,
      bv, 1.0f, vth, nullptr, T_SEQ, sXD, 1, D_MODEL, T_SEQ);  // store V^T

  // scores = q @ k^T / sqrt(K): per batch M=N=T, Kdim=KV_DIM
  dim3 gScore(T_SEQ / 128, T_SEQ / 256, N_BATCH);
  attn_gemm_wmma<<<gScore, 256, 0, stream>>>(qh, KV_DIM, sXD, kh, KV_DIM, sXD,
      nullptr, 0.03125f, S, nullptr, T_SEQ, sTT, 0, KV_DIM, T_SEQ);

  attn_softmax_rows<<<(int)(N_BATCH * (long)T_SEQ), 256, 0, stream>>>(S);

  // out = P @ v: per batch M=T, N=V_DIM, Kdim=T; f32 output
  dim3 gOut(KV_DIM / 128, T_SEQ / 256, N_BATCH);
  attn_gemm_wmma<<<gOut, 256, 0, stream>>>(S, T_SEQ, sTT, vth, T_SEQ, sXD,
      nullptr, 1.0f, nullptr, out, KV_DIM, sXD, 0, T_SEQ, T_SEQ);
}
